// GCN_18863496364059
// MI455X (gfx1250) — matrix operand order repaired and checked by reading, not compiled, not run on hardware
//
#include <hip/hip_runtime.h>
#include <hip/hip_bf16.h>

// ---------------------------------------------------------------------------
// GCN FiLM + segment-mean, fused for MI455X (gfx1250, wave32, WMMA).
//
//   h = relu(pose @ W1 + b1)                  (E x 1280)
//   e = sigmoid(h @ W2 + b2)                  (E x 2560)  <- bf16x3 WMMA
//   m = gamma * image[src] + beta             fused epilogue
//   out = segment_mean(m, dst)                f32 atomics into L2-resident out
// ---------------------------------------------------------------------------

typedef __attribute__((ext_vector_type(16))) __bf16 v16bf;
typedef __attribute__((ext_vector_type(8)))  float  v8f;
typedef int v4i __attribute__((vector_size(16)));

#define C_CH   1280
#define TWO_C  2560
#define NKT    40            // K tiles of 32  (1280/32)
#define NNT    160           // N tiles of 16  (2560/16)
#define EPB    32            // edges per block (two M=16 tiles)
#define HW     16            // 4x4 spatial

#if defined(__AMDGCN__) && __has_builtin(__builtin_amdgcn_global_load_async_to_lds_b32)
#define USE_ASYNC_LDS 1
#endif

// ---------------------------------------------------------------- utilities
__device__ __forceinline__ void split_bf16(float x, __bf16& hi, __bf16& lo) {
    hi = (__bf16)x;
    lo = (__bf16)(x - (float)hi);
}

// ------------------------------------------------------------- zero buffers
__global__ void zero_kernel(float* __restrict__ out, float* __restrict__ cnt,
                            int n_out, int n_nodes) {
    int stride = gridDim.x * blockDim.x;
    for (int i = blockIdx.x * blockDim.x + threadIdx.x; i < n_out; i += stride)
        out[i] = 0.0f;
    int t = blockIdx.x * blockDim.x + threadIdx.x;
    if (t < n_nodes) cnt[t] = 0.0f;
}

// ------------------------------------------------------------ degree counts
__global__ void count_kernel(const int* __restrict__ dst, float* __restrict__ cnt,
                             int E) {
    int e = blockIdx.x * blockDim.x + threadIdx.x;
    if (e < E) atomicAdd(&cnt[dst[e]], 1.0f);
}

// ------------------------------------------------------------------ pack W2
// Split W2 (f32, 1280x2560 row-major) into bf16 hi/lo, pre-swizzled into the
// WMMA B-fragment register layout:
//   fragment(nt,kt) = 1024 bf16 (512 hi, 512 lo)
//   lane = (N%16) + 16*((K%16)>=8),  element e = K%8 + 8*((K%32)>=16)  (table k order)
__global__ void pack_w2_kernel(const float* __restrict__ W2,
                               __bf16* __restrict__ wsB) {
    int idx = blockIdx.x * blockDim.x + threadIdx.x;   // over 1280*2560
    if (idx >= C_CH * TWO_C) return;
    int k = idx / TWO_C;          // row of W2 (K)
    int n = idx - k * TWO_C;      // col of W2 (N)
    float v = W2[idx];            // fully coalesced read
    __bf16 hi, lo; split_bf16(v, hi, lo);
    int nt = n >> 4;
    int kt = k >> 5;
    int kk = k & 31;
    int lane = (n & 15) + (((kk >> 3) & 1) << 4);
    int e    = (kk & 7) + ((kk >> 4) << 3);
    size_t base = (size_t)(nt * NKT + kt) * 1024;
    wsB[base +       lane * 16 + e] = hi;
    wsB[base + 512 + lane * 16 + e] = lo;
}

// ------------------------------------------------- fused GEMM + FiLM scatter
// One block = 32 edges. Dynamic LDS holds h split hi/lo in WMMA A layout:
//   A-frag(mt,kt): 32 lanes x 16 bf16, lane = M + 16*half,
//   element 2j+pos maps K per the CDNA5 16-bit A table.
__device__ __forceinline__ int a_swizzle_elem(int k32) {
    int pos = k32 & 1;
    int t   = k32 >> 1;
    int j   = (t & 3) + ((t & 8) >> 1);   // t 0-3->j=t h0 ; 4-7->j h1 ; 8-11->j+4 h0 ; 12-15->j+4 h1
    return 2 * j + pos;
}
__device__ __forceinline__ int a_swizzle_half(int k32) {
    return ((k32 >> 1) >> 2) & 1;
}

__global__ void __launch_bounds__(256)
gemm_film_scatter(const float* __restrict__ pose,
                  const float* __restrict__ image,
                  const float* __restrict__ W1,
                  const float* __restrict__ b1,
                  const float* __restrict__ b2,
                  const int*   __restrict__ src,
                  const int*   __restrict__ dst,
                  const __bf16* __restrict__ wsB,
                  float* __restrict__ out) {
    extern __shared__ char smem[];
    __bf16* hHi  = (__bf16*)smem;                 // 2*40*512 = 40960 elems (80 KB)
    __bf16* hLo  = hHi + 2 * NKT * 512;           // 80 KB
    float*  poseS = (float*)(hLo + 2 * NKT * 512);// 32*12 f32
    int*    sdS   = (int*)(poseS + 32 * 12);      // src[32] ++ dst[32]

    const int tid = threadIdx.x;
    const int e0  = blockIdx.x * EPB;

    // ---- stage pose rows + src/dst via ASYNCcnt global->LDS copies -------
#ifdef USE_ASYNC_LDS
    if (tid < EPB * 9) {
        int e = tid / 9, k = tid - e * 9;
        __builtin_amdgcn_global_load_async_to_lds_b32(
            (int*)(pose + (size_t)(e0 + e) * 9 + k),
            (int*)(poseS + e * 12 + k), 0, 0);
    }
    if (tid < EPB / 4) {        // 32 ints of src/dst as 8 lanes x b128
        __builtin_amdgcn_global_load_async_to_lds_b128(
            (v4i*)(src + e0 + tid * 4), (v4i*)(sdS + tid * 4), 0, 0);
        __builtin_amdgcn_global_load_async_to_lds_b128(
            (v4i*)(dst + e0 + tid * 4), (v4i*)(sdS + EPB + tid * 4), 0, 0);
    }
#if __has_builtin(__builtin_amdgcn_s_wait_asynccnt)
    __builtin_amdgcn_s_wait_asynccnt(0);
#else
    asm volatile("s_wait_asynccnt 0x0" ::: "memory");
#endif
#else
    if (tid < EPB * 9) {
        int e = tid / 9, k = tid - e * 9;
        poseS[e * 12 + k] = pose[(size_t)(e0 + e) * 9 + k];
    }
    if (tid < EPB) {
        sdS[tid]       = src[e0 + tid];
        sdS[EPB + tid] = dst[e0 + tid];
    }
#endif
    __syncthreads();

    // ---- phase 1: h = relu(pose@W1+b1), bf16-split, swizzled into LDS ----
    for (int i = 0; i < (EPB * C_CH) / 256; ++i) {
        int flat = i * 256 + tid;                 // 1280 | 256 -> one edge per chunk
        int el   = flat / C_CH;
        int c    = flat - el * C_CH;
        float acc = b1[c];
        #pragma unroll
        for (int k = 0; k < 9; ++k)
            acc = fmaf(poseS[el * 12 + k], W1[k * C_CH + c], acc);
        float h = fmaxf(acc, 0.0f);
        __bf16 hi, lo; split_bf16(h, hi, lo);
        int mt  = el >> 4, M = el & 15;
        int kt  = c >> 5,  k32 = c & 31;
        int lane = M + (a_swizzle_half(k32) << 4);
        int idx  = ((mt * NKT + kt) * 32 + lane) * 16 + a_swizzle_elem(k32);
        hHi[idx] = hi;
        hLo[idx] = lo;
    }
    __syncthreads();

    // ---- phase 2: per-wave GEMM over 20 N-tiles, bf16x3, f32 accum -------
    const int wave = tid >> 5;
    const int lane = tid & 31;
    const int hf   = lane >> 4;

    for (int it = 0; it < NNT / 8; ++it) {
        const int nt = it * 8 + wave;
        const int n0 = nt * 16;
        v8f acc0 = {};    // edges 0..15 of block
        v8f acc1 = {};    // edges 16..31 of block
        const __bf16* bBase = wsB + (size_t)nt * NKT * 1024;

        for (int kt = 0; kt < NKT; ++kt) {
            const __bf16* bp = bBase + kt * 1024 + lane * 16;
            if (kt + 1 < NKT) __builtin_prefetch(bp + 1024, 0, 3);  // near-scope
            v16bf bhi = *(const v16bf*)bp;
            v16bf blo = *(const v16bf*)(bp + 512);
            v16bf a0h = *(const v16bf*)(hHi + ((0 * NKT + kt) * 32 + lane) * 16);
            v16bf a1h = *(const v16bf*)(hHi + ((1 * NKT + kt) * 32 + lane) * 16);
            v16bf a0l = *(const v16bf*)(hLo + ((0 * NKT + kt) * 32 + lane) * 16);
            v16bf a1l = *(const v16bf*)(hLo + ((1 * NKT + kt) * 32 + lane) * 16);
            // bf16x3: hi*hi + lo*hi + hi*lo  (f32 accumulate)
            acc0 = __builtin_amdgcn_wmma_f32_16x16x32_bf16(false, a0h, false, bhi, (short)0, acc0, false, false);
            acc0 = __builtin_amdgcn_wmma_f32_16x16x32_bf16(false, a0l, false, bhi, (short)0, acc0, false, false);
            acc0 = __builtin_amdgcn_wmma_f32_16x16x32_bf16(false, a0h, false, blo, (short)0, acc0, false, false);
            acc1 = __builtin_amdgcn_wmma_f32_16x16x32_bf16(false, a1h, false, bhi, (short)0, acc1, false, false);
            acc1 = __builtin_amdgcn_wmma_f32_16x16x32_bf16(false, a1l, false, bhi, (short)0, acc1, false, false);
            acc1 = __builtin_amdgcn_wmma_f32_16x16x32_bf16(false, a1h, false, blo, (short)0, acc1, false, false);
        }

        // ---- epilogue: sigmoid -> (gamma,beta) pair -> FiLM + atomic scatter
        const int   col  = n0 + (lane & 15);     // output column in [0,2560)
        const int   ch   = col >> 1;             // channel
        const float b2v  = b2[col];
        const int   hwb  = (lane & 1) * 8;       // even lane: hw 0-7, odd: 8-15

        #pragma unroll
        for (int mt = 0; mt < 2; ++mt) {
            v8f cc = (mt == 0) ? acc0 : acc1;
            #pragma unroll
            for (int r = 0; r < 8; ++r) {
                float z = cc[r] + b2v;
                float s = __frcp_rn(1.0f + __expf(-z));     // sigmoid
                float o = __shfl_xor(s, 1);                  // pair gamma/beta
                float gamma = (lane & 1) ? o : s;
                float beta  = (lane & 1) ? s : o;
                int el = mt * 16 + r + 8 * hf;               // D layout: M = r + 8*(lane/16)
                int sN = sdS[el];
                int dN = sdS[EPB + el];
                const float4* ip = (const float4*)(image + ((size_t)sN * C_CH + ch) * HW + hwb);
                float*        op = out + ((size_t)dN * C_CH + ch) * HW + hwb;
                float4 v0 = ip[0];
                float4 v1 = ip[1];
                atomicAdd(op + 0, fmaf(gamma, v0.x, beta));
                atomicAdd(op + 1, fmaf(gamma, v0.y, beta));
                atomicAdd(op + 2, fmaf(gamma, v0.z, beta));
                atomicAdd(op + 3, fmaf(gamma, v0.w, beta));
                atomicAdd(op + 4, fmaf(gamma, v1.x, beta));
                atomicAdd(op + 5, fmaf(gamma, v1.y, beta));
                atomicAdd(op + 6, fmaf(gamma, v1.z, beta));
                atomicAdd(op + 7, fmaf(gamma, v1.w, beta));
            }
        }
    }
}

// ----------------------------------------------------------- segment mean
__global__ void normalize_kernel(float* __restrict__ out,
                                 const float* __restrict__ cnt, int n_out) {
    int stride = gridDim.x * blockDim.x;
    for (int i = blockIdx.x * blockDim.x + threadIdx.x; i < n_out; i += stride) {
        int node = i / (C_CH * HW);
        float c = cnt[node];
        out[i] = (c > 0.0f) ? out[i] / c : 0.0f;
    }
}

// ---------------------------------------------------------------------------
extern "C" void kernel_launch(void* const* d_in, const int* in_sizes, int n_in,
                              void* d_out, int out_size, void* d_ws, size_t ws_size,
                              hipStream_t stream) {
    const float* pose  = (const float*)d_in[0];
    const float* image = (const float*)d_in[1];
    const float* W1    = (const float*)d_in[2];
    const float* b1    = (const float*)d_in[3];
    const float* W2    = (const float*)d_in[4];
    const float* b2    = (const float*)d_in[5];
    const int*   src   = (const int*)d_in[6];
    const int*   dst   = (const int*)d_in[7];
    float*       out   = (float*)d_out;

    const int E      = in_sizes[6];                    // 8192
    const int nNodes = in_sizes[1] / (C_CH * HW);      // 1024

    // workspace: packed W2 fragments (13.1 MB) ++ degree counts
    __bf16* wsB = (__bf16*)d_ws;
    float*  cnt = (float*)((char*)d_ws + (size_t)NNT * NKT * 1024 * sizeof(__bf16));

    const size_t smem = (size_t)(2 * NKT * 512) * 2 * sizeof(__bf16)  // h hi+lo
                      + 32 * 12 * sizeof(float)                       // pose
                      + 2 * EPB * sizeof(int);                        // src/dst
    (void)hipFuncSetAttribute((const void*)gemm_film_scatter,
                              hipFuncAttributeMaxDynamicSharedMemorySize,
                              (int)smem);

    zero_kernel<<<2048, 256, 0, stream>>>(out, cnt, out_size, nNodes);
    count_kernel<<<(E + 255) / 256, 256, 0, stream>>>(dst, cnt, E);
    pack_w2_kernel<<<(C_CH * TWO_C + 255) / 256, 256, 0, stream>>>(W2, wsB);
    gemm_film_scatter<<<E / EPB, 256, smem, stream>>>(pose, image, W1, b1, b2,
                                                      src, dst, wsB, out);
    normalize_kernel<<<2048, 256, 0, stream>>>(out, cnt, out_size);
}
